// CausalSelfAttention_21157008900511
// MI455X (gfx1250) — matrix operand order repaired and checked by reading, not compiled, not run on hardware
//
#include <hip/hip_runtime.h>
#include <hip/hip_bf16.h>

typedef _Float16 half_t;
typedef __attribute__((ext_vector_type(16))) _Float16 v16h;
typedef __attribute__((ext_vector_type(8)))  _Float16 v8h;
typedef __attribute__((ext_vector_type(8)))  float    v8f;
typedef __attribute__((ext_vector_type(4)))  unsigned int u32x4;
typedef __attribute__((ext_vector_type(8)))  int          i32x8;
typedef __attribute__((ext_vector_type(4)))  int          i32x4;

__device__ __forceinline__ v16h ld16(const half_t* p0, const half_t* p1) {
    v8h lo = *(const v8h*)p0;
    v8h hi = *(const v8h*)p1;
    return __builtin_shufflevector(lo, hi, 0,1,2,3,4,5,6,7,8,9,10,11,12,13,14,15);
}

__device__ __forceinline__ v8f wmma16(v16h a, v16h b, v8f c) {
    return __builtin_amdgcn_wmma_f32_16x16x32_f16(false, a, false, b, (short)0, c, false, false);
}

// ---------------------------------------------------------------------------
// TDM: 2D tile load global -> LDS (row-major), optional per-row LDS padding.
//   tile_x      : elements per row (2-byte elements)
//   tile_y      : rows
//   stride_x    : row stride in global memory (elements)
//   pad_interval: ISA encoding (3 -> every 16 DWORDs, 4 -> every 32 DWORDs)
//   pad_amount  : ISA encoding (3 -> 4 DWORDs = 8 halves)
// One wave issues this; TENSORcnt tracks completion.
// ---------------------------------------------------------------------------
__device__ __forceinline__ void tdm_load_2d_f16(unsigned int lds_addr,
                                                const half_t* gptr,
                                                unsigned int tile_x,
                                                unsigned int tile_y,
                                                unsigned int stride_x,
                                                unsigned int pad_interval,
                                                unsigned int pad_amount) {
    unsigned long long ga = (unsigned long long)(__SIZE_TYPE__)gptr;
    u32x4 g0;
    g0[0] = 1u;                                              // count=1, no gather
    g0[1] = lds_addr;                                        // lds_addr[31:0]
    g0[2] = (unsigned int)ga;                                // global_addr[31:0]
    g0[3] = ((unsigned int)(ga >> 32) & 0x01FFFFFFu)         // global_addr[56:32]
          | (2u << 30);                                      // type = 2 ("image")
    i32x8 g1;
    g1[0] = (int)((pad_amount << 25) | (pad_interval << 22)
          | (1u << 20)                                       // pad_enable
          | (1u << 16));                                     // data_size = 2B
    g1[1] = (int)(tile_x << 16);                             // tensor_dim0[15:0]
    g1[2] = (int)(tile_y << 16);                             // tensor_dim1[15:0]
    g1[3] = (int)(tile_x << 16);                             // tile_dim0
    g1[4] = (int)tile_y;                                     // tile_dim1 (tile_dim2=0)
    g1[5] = (int)stride_x;                                   // tensor_dim0_stride lo32
    g1[6] = 0;
    g1[7] = 0;
    i32x4 z4 = {0, 0, 0, 0};
#if defined(__clang_major__) && __clang_major__ >= 23
    i32x8 z8 = {0, 0, 0, 0, 0, 0, 0, 0};
    __builtin_amdgcn_tensor_load_to_lds(g0, g1, z4, z4, z8, 0);
#else
    __builtin_amdgcn_tensor_load_to_lds(g0, g1, z4, z4, 0);
#endif
}

// ---------------------------------------------------------------------------
// f32 -> f16 conversion
// ---------------------------------------------------------------------------
__global__ void cvt_f32_to_f16(const float* __restrict__ in, half_t* __restrict__ out, int n) {
    int i = blockIdx.x * blockDim.x + threadIdx.x;
    if (i < n) out[i] = (half_t)in[i];
}

// ---------------------------------------------------------------------------
// Tiled WMMA GEMM: out[M,N] = A[M,K] * B[K,N], f16 inputs, f16 or f32 out.
// Block 256 threads = 8 waves; block tile 128x128; wave tile 64x32; BK=32.
// A tile staged by the Tensor Data Mover; B tile transposed cooperatively.
// ---------------------------------------------------------------------------
template <bool STORE_F32>
__global__ void __launch_bounds__(256)
gemm_f16_wmma(const half_t* __restrict__ A, const half_t* __restrict__ Bm,
              void* __restrict__ out, int M, int N, int K) {
    constexpr int BM = 128, BN = 128, BK = 32;
    constexpr int AST = BK + 8;   // 40 halves (TDM pad: 8 halves per 32-half row)
    constexpr int BST = BK + 8;   // transposed B tile: [n][k]
    __shared__ half_t sA [BM * AST];
    __shared__ half_t sBt[BN * BST];

    const int t    = threadIdx.x;
    const int lane = t & 31;
    const int w    = t >> 5;           // wave 0..7
    const int wr   = w & 1;            // row group (2 x 64 rows)
    const int wc   = w >> 1;           // col group (4 x 32 cols)
    const int hi   = lane >> 4;        // lane half
    const int l15  = lane & 15;
    const int bm   = blockIdx.y * BM;
    const int bn   = blockIdx.x * BN;

    const unsigned int sA_lds = (unsigned int)(__SIZE_TYPE__)&sA[0];

    v8f acc[4][2] = {};

    for (int k0 = 0; k0 < K; k0 += BK) {
        // --- A tile 128x32 via TDM (wave 0 issues; pad -> 40-half rows) ---
        if (t < 32)
            tdm_load_2d_f16(sA_lds, A + (size_t)bm * K + k0,
                            BK, BM, (unsigned int)K,
                            /*pad_interval=*/3 /*16 DWORDs*/, /*pad_amount=*/3);

        // --- stage B tile 32x128 transposed into sBt[n][k] ---
        {
            int kk = t >> 3;
            int n0 = (t & 7) * 16;
            const half_t* gp = Bm + (size_t)(k0 + kk) * N + bn + n0;
            v8h b0 = *(const v8h*)(gp);
            v8h b1 = *(const v8h*)(gp + 8);
#pragma unroll
            for (int j = 0; j < 8; ++j) sBt[(n0 + j)     * BST + kk] = b0[j];
#pragma unroll
            for (int j = 0; j < 8; ++j) sBt[(n0 + 8 + j) * BST + kk] = b1[j];
            if (k0 + BK < K)  // prefetch next B tile rows into L2
                __builtin_prefetch(Bm + (size_t)(k0 + BK + kk) * N + bn + n0, 0, 1);
        }
        if (t < 32) __builtin_amdgcn_s_wait_tensorcnt(0);
        __syncthreads();

        // --- fragments + 8 WMMAs per wave ---
        v16h af[4], bf[2];
#pragma unroll
        for (int rt = 0; rt < 4; ++rt) {
            const half_t* p = &sA[(wr * 64 + rt * 16 + l15) * AST + hi * 8];
            af[rt] = ld16(p, p + 16);              // K chunks {hi*8..+7, 16+hi*8..+7}
        }
#pragma unroll
        for (int ct = 0; ct < 2; ++ct) {
            const half_t* p = &sBt[(wc * 32 + ct * 16 + l15) * BST + hi * 8];
            bf[ct] = ld16(p, p + 16);              // K chunks {hi*8..+7, 16+hi*8..+7}
        }
#pragma unroll
        for (int rt = 0; rt < 4; ++rt)
#pragma unroll
            for (int ct = 0; ct < 2; ++ct)
                acc[rt][ct] = wmma16(af[rt], bf[ct], acc[rt][ct]);
        __syncthreads();
    }

    // --- epilogue: C layout element (i,lane) -> row i+8*hi, col l15 ---
#pragma unroll
    for (int rt = 0; rt < 4; ++rt) {
#pragma unroll
        for (int ct = 0; ct < 2; ++ct) {
            int gn = bn + wc * 32 + ct * 16 + l15;
#pragma unroll
            for (int i = 0; i < 8; ++i) {
                int gm = bm + wr * 64 + rt * 16 + i + hi * 8;
                if constexpr (STORE_F32)
                    ((float*)out)[(size_t)gm * N + gn] = acc[rt][ct][i];
                else
                    ((half_t*)out)[(size_t)gm * N + gn] = (half_t)acc[rt][ct][i];
            }
        }
    }
}

// ---------------------------------------------------------------------------
// Flash attention: one workgroup per (b, h, 64-query tile); 4 waves x 16 q.
// qkv: f16 [B, T, 3C] (q | k | v per row).  Y: f16 [B, T, C].
// Q and K tiles staged by TDM; V transposed cooperatively.
// ---------------------------------------------------------------------------
__global__ void __launch_bounds__(128)
attn_flash_wmma(const half_t* __restrict__ qkv, half_t* __restrict__ Y) {
    constexpr int T = 2048, NH = 16, HD = 64, C = NH * HD, C3 = 3 * C;
    constexpr int ST = HD + 8;                     // 72-half row stride (TDM pad)
    __shared__ half_t sQ [64 * ST];                // [q][hd]
    __shared__ half_t sK [64 * ST];                // [key][hd]
    __shared__ half_t sVt[HD * ST];                // [hd][key]  (transposed V)
    __shared__ half_t sP [4][16 * ST];             // per-wave P tile [q][key]

    const int t    = threadIdx.x;
    const int lane = t & 31;
    const int w    = t >> 5;                       // wave 0..3 -> query rows w*16..
    const int hi   = lane >> 4;
    const int l15  = lane & 15;
    const int qb   = blockIdx.x;                   // query block (64 queries)
    const int h    = blockIdx.y;
    const int b    = blockIdx.z;
    const int q0   = qb * 64;

    const size_t baseQ = (size_t)b * T * C3 + (size_t)h * HD;
    const size_t baseK = baseQ + C;
    const size_t baseV = baseQ + 2 * C;

    // stage Q tile 64x64 via TDM (row = 64 halves = 32 DWORDs -> pad_interval 4)
    if (t < 32) {
        tdm_load_2d_f16((unsigned int)(__SIZE_TYPE__)&sQ[0],
                        qkv + baseQ + (size_t)q0 * C3,
                        HD, 64, C3, /*pad_interval=*/4, /*pad_amount=*/3);
        __builtin_amdgcn_s_wait_tensorcnt(0);
    }
    __syncthreads();

    v16h qf[2];
#pragma unroll
    for (int hc = 0; hc < 2; ++hc) {
        const half_t* p = &sQ[(w * 16 + l15) * ST + hc * 32 + hi * 8];
        qf[hc] = ld16(p, p + 16);
    }

    float mrow[8], lrow[8];
#pragma unroll
    for (int i = 0; i < 8; ++i) { mrow[i] = -1e30f; lrow[i] = 0.0f; }
    v8f oacc[4] = {};

    const float scale = 0.125f;                    // 1/sqrt(64)

    for (int kb = 0; kb <= qb; ++kb) {
        __syncthreads();                           // WAR on sK/sVt across waves
        // K tile via TDM; V tile transposed cooperatively
        if (t < 32)
            tdm_load_2d_f16((unsigned int)(__SIZE_TYPE__)&sK[0],
                            qkv + baseK + (size_t)(kb * 64) * C3,
                            HD, 64, C3, /*pad_interval=*/4, /*pad_amount=*/3);
        {
            int r = t >> 1, c0 = (t & 1) * 32;
            const half_t* gv = qkv + baseV + (size_t)(kb * 64 + r) * C3 + c0;
#pragma unroll
            for (int j = 0; j < 32; ++j)
                sVt[(c0 + j) * ST + r] = gv[j];
        }
        if (t < 32) __builtin_amdgcn_s_wait_tensorcnt(0);
        __syncthreads();

        // S = Q * K^T : 4 key tiles x 2 K-chunks over HD
        v8f s[4] = {};
#pragma unroll
        for (int ct = 0; ct < 4; ++ct) {
#pragma unroll
            for (int hc = 0; hc < 2; ++hc) {
                const half_t* p = &sK[(ct * 16 + l15) * ST + hc * 32 + hi * 8];
                v16h kf = ld16(p, p + 16);
                s[ct] = wmma16(qf[hc], kf, s[ct]);
            }
        }

        // scale + causal mask (only diagonal block needs masking)
        const bool diag = (kb == qb);
#pragma unroll
        for (int ct = 0; ct < 4; ++ct) {
#pragma unroll
            for (int i = 0; i < 8; ++i) {
                float v = s[ct][i] * scale;
                if (diag) {
                    int kg = kb * 64 + ct * 16 + l15;
                    int qg = q0 + w * 16 + i + hi * 8;
                    if (kg > qg) v = -1e30f;
                }
                s[ct][i] = v;
            }
        }

        // online softmax: row max across 4 tiles + 16-lane half-wave reduce
        float mx[8], alpha[8], rs[8];
#pragma unroll
        for (int i = 0; i < 8; ++i) {
            float m = s[0][i];
            m = fmaxf(m, s[1][i]); m = fmaxf(m, s[2][i]); m = fmaxf(m, s[3][i]);
#pragma unroll
            for (int d = 1; d < 16; d <<= 1) m = fmaxf(m, __shfl_xor(m, d, 32));
            float mn = fmaxf(mrow[i], m);
            alpha[i] = __expf(mrow[i] - mn);
            mrow[i]  = mn;
            rs[i]    = 0.0f;
        }
#pragma unroll
        for (int ct = 0; ct < 4; ++ct)
#pragma unroll
            for (int i = 0; i < 8; ++i) {
                float p = __expf(s[ct][i] - mrow[i]);
                s[ct][i] = p;
                rs[i] += p;
            }
#pragma unroll
        for (int i = 0; i < 8; ++i) {
#pragma unroll
            for (int d = 1; d < 16; d <<= 1) rs[i] += __shfl_xor(rs[i], d, 32);
            lrow[i] = lrow[i] * alpha[i] + rs[i];
        }
#pragma unroll
        for (int ct = 0; ct < 4; ++ct)
#pragma unroll
            for (int i = 0; i < 8; ++i) oacc[ct][i] *= alpha[i];

        // re-stage P (C layout) -> per-wave LDS -> A-fragment layout
#pragma unroll
        for (int ct = 0; ct < 4; ++ct)
#pragma unroll
            for (int i = 0; i < 8; ++i)
                sP[w][(i + hi * 8) * ST + ct * 16 + l15] = (half_t)s[ct][i];

        // O += P * V : 4 hd tiles x 2 K-chunks over keys
#pragma unroll
        for (int ct = 0; ct < 4; ++ct) {
#pragma unroll
            for (int kc = 0; kc < 2; ++kc) {
                const half_t* pa = &sP[w][l15 * ST + kc * 32 + hi * 8];
                v16h pf = ld16(pa, pa + 16);
                const half_t* pb = &sVt[(ct * 16 + l15) * ST + kc * 32 + hi * 8];
                v16h vf = ld16(pb, pb + 16);
                oacc[ct] = wmma16(pf, vf, oacc[ct]);
            }
        }
    }

    // normalize + store y (f16) in [B, T, C] layout
#pragma unroll
    for (int ct = 0; ct < 4; ++ct) {
#pragma unroll
        for (int i = 0; i < 8; ++i) {
            int q  = q0 + w * 16 + i + hi * 8;
            int hd = ct * 16 + l15;
            float o = oacc[ct][i] / lrow[i];
            Y[(size_t)(b * T + q) * C + h * HD + hd] = (half_t)o;
        }
    }
}

// ---------------------------------------------------------------------------
// launch
// ---------------------------------------------------------------------------
extern "C" void kernel_launch(void* const* d_in, const int* in_sizes, int n_in,
                              void* d_out, int out_size, void* d_ws, size_t ws_size,
                              hipStream_t stream) {
    constexpr int Bc = 2, T = 2048, C = 1024;
    const float* x      = (const float*)d_in[0];
    const float* w_attn = (const float*)d_in[1];
    const float* w_proj = (const float*)d_in[2];
    float* out = (float*)d_out;

    const size_t nx  = (size_t)Bc * T * C;       // 4,194,304
    const size_t nwa = (size_t)C * 3 * C;        // 3,145,728
    const size_t nwp = (size_t)C * C;            // 1,048,576
    const size_t nqkv = (size_t)Bc * T * 3 * C;  // 12,582,912

    half_t* ws    = (half_t*)d_ws;
    half_t* x_h   = ws;
    half_t* wa_h  = x_h  + nx;
    half_t* wp_h  = wa_h + nwa;
    half_t* qkv_h = wp_h + nwp;
    half_t* y_h   = qkv_h + nqkv;                // total ~48 MB of d_ws

    cvt_f32_to_f16<<<(int)((nx  + 255) / 256), 256, 0, stream>>>(x,      x_h,  (int)nx);
    cvt_f32_to_f16<<<(int)((nwa + 255) / 256), 256, 0, stream>>>(w_attn, wa_h, (int)nwa);
    cvt_f32_to_f16<<<(int)((nwp + 255) / 256), 256, 0, stream>>>(w_proj, wp_h, (int)nwp);

    // qkv = x @ w_attn : [4096,1024] x [1024,3072] -> f16
    gemm_f16_wmma<false><<<dim3(3072 / 128, 4096 / 128), 256, 0, stream>>>(
        x_h, wa_h, (void*)qkv_h, Bc * T, 3 * C, C);

    // flash attention -> y f16 [B,T,C]
    attn_flash_wmma<<<dim3(T / 64, 16, Bc), 128, 0, stream>>>(qkv_h, y_h);

    // out = y @ w_proj : [4096,1024] x [1024,1024] -> f32 d_out
    gemm_f16_wmma<true><<<dim3(1024 / 128, 4096 / 128), 256, 0, stream>>>(
        y_h, wp_h, (void*)out, Bc * T, C, C);
}
